// Conv1D_LSTM_SelfMultiHeadAttention_48850958024754
// MI455X (gfx1250) — matrix operand order repaired and checked by reading, not compiled, not run on hardware
//
#include <hip/hip_runtime.h>
#include <hip/hip_bf16.h>
#include <math.h>

// ---------------------------------------------------------------------------
// MI455X (gfx1250) implementation. wave32, WMMA f32<=f16 16x16x32, TDM staging.
// Shapes: B=64, C=32, L=1549 -> conv(pad1,k16)=1536 -> pool3 -> S=512
// BiLSTM x2 (H=256, D2=512), MHA (NH=8, DK=64), LN, mean-pool, FC -> (64,1)
// ---------------------------------------------------------------------------

typedef __attribute__((ext_vector_type(16))) _Float16 v16h;
typedef __attribute__((ext_vector_type(8)))  _Float16 v8h;
typedef __attribute__((ext_vector_type(8)))  float    v8f;
typedef __attribute__((ext_vector_type(4)))  unsigned int v4u;
typedef __attribute__((ext_vector_type(8)))  int v8i;
typedef __attribute__((ext_vector_type(4)))  int v4i;

#define WMMA_F16(a, b, c) \
  __builtin_amdgcn_wmma_f32_16x16x32_f16(false, (a), false, (b), (short)0, (c), false, false)

#if defined(__has_builtin)
#if __has_builtin(__builtin_amdgcn_tensor_load_to_lds) && \
    __has_builtin(__builtin_amdgcn_s_wait_tensorcnt)
#define HAVE_TDM 1
#endif
#endif
#ifndef HAVE_TDM
#define HAVE_TDM 0
#endif

static constexpr int Bsz = 64;
static constexpr int Cch = 32;
static constexpr int Lin = 1549;
static constexpr int S   = 512;
static constexpr int HID = 256;
static constexpr int D2  = 512;
static constexpr int NH  = 8;
static constexpr int DK  = 64;

__device__ __forceinline__ float sigf(float x) { return 1.f / (1.f + __expf(-x)); }

#if HAVE_TDM
// ---------------------------------------------------------------------------
// TDM: 1-D tile (nelem u16 elements, contiguous in memory) -> LDS, inserting
// 1 DWORD of padding every 32 DWORDs (row stride 128B -> 132B) so subsequent
// K-major LDS gathers step 33 banks instead of re-hitting bank 0.
// D# built per CDNA5 ISA 8.3/8.4 (group0: count/lds/global/type, group1:
// data_size=2B, pad_enable, pad_interval=4 (32 DW), dims; groups 2/3 = 0).
// ---------------------------------------------------------------------------
__device__ __forceinline__ void tdm_load_1d_padded(unsigned lds_off,
                                                   unsigned long long gaddr,
                                                   int nelem_u16) {
  v4u g0;
  g0[0] = 1u;                                    // count=1, user descriptor
  g0[1] = lds_off;                               // lds_addr (bytes)
  g0[2] = (unsigned)gaddr;                       // global_addr[31:0]
  g0[3] = (unsigned)((gaddr >> 32) & 0x1FFFFFFu) // global_addr[56:32]
          | (2u << 30);                          // type=2 ("image")
  v8i g1;
  g1[0] = (int)((1u << 16)        // data_size = 2 bytes
                | (1u << 20)      // pad_enable
                | (4u << 22));    // pad_interval: 32 DWORDs (pad_amount=0 -> 1 DWORD)
  g1[1] = (int)(((unsigned)nelem_u16 & 0xffffu) << 16);  // tensor_dim0[15:0]
  g1[2] = (int)(((unsigned)nelem_u16 >> 16) | (1u << 16)); // td0[31:16] | tensor_dim1=1
  g1[3] = (int)(((unsigned)nelem_u16 & 0xffffu) << 16);  // tile_dim0
  g1[4] = 1;                                             // tile_dim1 = 1
  g1[5] = nelem_u16;                                     // tensor_dim0_stride lo32
  g1[6] = 0;
  g1[7] = 0;
  v4i g2 = {0, 0, 0, 0};
  v4i g3 = {0, 0, 0, 0};
#if defined(__clang_major__) && (__clang_major__ >= 23)
  v8i gx = {0, 0, 0, 0, 0, 0, 0, 0};
  __builtin_amdgcn_tensor_load_to_lds(g0, g1, g2, g3, gx, 0);
#else
  __builtin_amdgcn_tensor_load_to_lds(g0, g1, g2, g3, 0);
#endif
}
#endif  // HAVE_TDM

// ---------------------------------------------------------------------------
// conv1d + BN + ReLU + maxpool3, output transposed to (B,S,32) f16
// ---------------------------------------------------------------------------
__launch_bounds__(32)
__global__ void conv_bn_pool(const float* __restrict__ x, const float* __restrict__ w,
                             const float* __restrict__ cb, const float* __restrict__ gm,
                             const float* __restrict__ bt, const float* __restrict__ mn,
                             const float* __restrict__ vr, _Float16* __restrict__ out) {
  const int b = blockIdx.x >> 9;
  const int s = blockIdx.x & 511;
  const int co = threadIdx.x;
  __shared__ float xs[32][18];
  const int base = 3 * s - 1;  // input idx for pool pos p, tap k is base+p+k
  for (int i = co; i < 32 * 18; i += 32) {
    const int ci = i / 18, off = i % 18, idx = base + off;
    xs[ci][off] = (idx >= 0 && idx < Lin) ? x[((size_t)b * 32 + ci) * Lin + idx] : 0.f;
  }
  __syncthreads();
  const float isv = rsqrtf(vr[co] + 1e-5f) * gm[co];
  float m = -1e30f;
  for (int p = 0; p < 3; ++p) {
    float acc = cb[co];
    for (int ci = 0; ci < 32; ++ci) {
      const float* wr = w + ((size_t)co * 32 + ci) * 16;
#pragma unroll
      for (int k = 0; k < 16; ++k) acc += xs[ci][p + k] * wr[k];
    }
    acc = (acc - mn[co]) * isv + bt[co];
    acc = fmaxf(acc, 0.f);
    m = fmaxf(m, acc);
  }
  out[((size_t)b * S + s) * 32 + co] = (_Float16)m;
}

// ---------------------------------------------------------------------------
// weight packing: Wc = [W_ih | W_hh] as f16, row-major (1024 x (Din+256))
// ---------------------------------------------------------------------------
__global__ void pack_wc(const float* __restrict__ Wih, const float* __restrict__ Whh,
                        _Float16* __restrict__ Wc, int Din) {
  const int K = Din + 256;
  const size_t tot = (size_t)1024 * K;
  for (size_t i = blockIdx.x * (size_t)blockDim.x + threadIdx.x; i < tot;
       i += (size_t)gridDim.x * blockDim.x) {
    const int row = (int)(i / K), k = (int)(i % K);
    const float v = (k < Din) ? Wih[(size_t)row * Din + k] : Whh[(size_t)row * 256 + (k - Din)];
    Wc[i] = (_Float16)v;
  }
}

__global__ void cvt_f16(const float* __restrict__ src, _Float16* __restrict__ dst, int n) {
  const int i = blockIdx.x * 256 + threadIdx.x;
  if (i < n) dst[i] = (_Float16)src[i];
}

// ---------------------------------------------------------------------------
// Persistent single-workgroup LSTM direction. 1024 thr = 32 waves.
// Per step: gates(64x1024) = [x_t | h](64xK) @ Wc^T, K = DIN+256, via WMMA.
// Wave w owns 2 tasks (16x16 batch x hidden blocks); owning all 4 gate tiles
// for a block means the cell update is wave-local; c stays in VGPRs.
// ---------------------------------------------------------------------------
template <int DIN>
__launch_bounds__(1024)
__global__ void lstm_dir(const _Float16* __restrict__ X, const _Float16* __restrict__ Wc,
                         const float* __restrict__ bias, _Float16* __restrict__ H16,
                         float* __restrict__ H32, int colOff, int rev) {
  constexpr int KTOT = DIN + HID;
  __shared__ __align__(16) _Float16 hbuf[64 * HID];  // 32 KB, f16 hidden state
  const int tid = threadIdx.x, wave = tid >> 5, lane = tid & 31;
  const int lrow = lane & 15, lg = lane >> 4;
  for (int i = tid; i < 64 * HID; i += 1024) hbuf[i] = (_Float16)0.f;
  float c[2][8];
#pragma unroll
  for (int tk = 0; tk < 2; ++tk)
#pragma unroll
    for (int j = 0; j < 8; ++j) c[tk][j] = 0.f;
  __syncthreads();

  for (int tt = 0; tt < S; ++tt) {
    const int t = rev ? (S - 1 - tt) : tt;
    v8f acc[2][4];
#pragma unroll
    for (int task = 0; task < 2; ++task) {
      const int id = wave + task * 32, nj = id & 15;
#pragma unroll
      for (int gi = 0; gi < 4; ++gi) {
        const float bv = bias[gi * HID + nj * 16 + lrow];
#pragma unroll
        for (int e = 0; e < 8; ++e) acc[task][gi][e] = bv;
      }
    }
#pragma unroll
    for (int task = 0; task < 2; ++task) {
      const int id = wave + task * 32, mi = id >> 4, nj = id & 15;
      const int brow = mi * 16 + lrow;
      const _Float16* xrow = X + ((size_t)brow * S + t) * DIN;
      {  // prefetch next timestep's activation row (global_prefetch_b8)
        const int tn = rev ? (t - 1) : (t + 1);
        if (tn >= 0 && tn < S) __builtin_prefetch(X + ((size_t)brow * S + tn) * DIN, 0, 1);
      }
      for (int kc = 0; kc < KTOT; kc += 32) {
        // A fragment (16x32 f16): rows = batch, K = [x_t | h]
        const int k0 = kc + lg * 16, k1 = kc + lg * 16 + 8;
        const v8h lo = (k0 < DIN) ? *(const v8h*)(xrow + k0)
                                  : *(const v8h*)(&hbuf[brow * HID + (k0 - DIN)]);
        const v8h hi = (k1 < DIN) ? *(const v8h*)(xrow + k1)
                                  : *(const v8h*)(&hbuf[brow * HID + (k1 - DIN)]);
        v16h a;
#pragma unroll
        for (int i = 0; i < 8; ++i) { a[i] = lo[i]; a[i + 8] = hi[i]; }
#pragma unroll
        for (int gi = 0; gi < 4; ++gi) {
          const _Float16* wp =
              Wc + (size_t)(gi * HID + nj * 16 + lrow) * KTOT + kc + lg * 16;
          const v8h bl = *(const v8h*)wp;
          const v8h bh = *(const v8h*)(wp + 8);
          v16h bf;
#pragma unroll
          for (int i = 0; i < 8; ++i) { bf[i] = bl[i]; bf[i + 8] = bh[i]; }
          acc[task][gi] = WMMA_F16(a, bf, acc[task][gi]);
        }
      }
    }
    __syncthreads();  // all reads of hbuf for step t done
#pragma unroll
    for (int task = 0; task < 2; ++task) {
      const int id = wave + task * 32, mi = id >> 4, nj = id & 15;
      const int hcol = nj * 16 + lrow;
#pragma unroll
      for (int j = 0; j < 8; ++j) {
        const int b = mi * 16 + j + 8 * lg;
        const float iv = sigf(acc[task][0][j]);
        const float fv = sigf(acc[task][1][j]);
        const float gv = tanhf(acc[task][2][j]);
        const float ov = sigf(acc[task][3][j]);
        const float cn = fv * c[task][j] + iv * gv;
        c[task][j] = cn;
        const float hv = ov * tanhf(cn);
        const _Float16 h16 = (_Float16)hv;
        hbuf[b * HID + hcol] = h16;
        H16[((size_t)b * S + t) * D2 + colOff + hcol] = h16;
        if (H32) H32[((size_t)b * S + t) * D2 + colOff + hcol] = hv;
      }
    }
    __syncthreads();  // new h visible before next step
    // NOP when not dispatched as a cluster; cluster-split variant syncs here.
    __builtin_amdgcn_s_cluster_barrier();
  }
}

// ---------------------------------------------------------------------------
// q/k/v projection: (BS x 512) @ W^T + bias, written as (B, NH, S, DK) f16
// one 16x16 tile per wave, 8 waves/block
// ---------------------------------------------------------------------------
__launch_bounds__(256)
__global__ void proj_gemm_qkv(const _Float16* __restrict__ A, const _Float16* __restrict__ W,
                              const float* __restrict__ bias, _Float16* __restrict__ Out) {
  const int tid = threadIdx.x, wave = tid >> 5, lane = tid & 31;
  const int lrow = lane & 15, lg = lane >> 4;
  const int id = blockIdx.x * 8 + wave;  // 65536 tiles total
  const int mi = id >> 5, ni = id & 31;
  const int n = ni * 16 + lrow;
  v8f acc;
  const float bv = bias[n];
#pragma unroll
  for (int e = 0; e < 8; ++e) acc[e] = bv;
  const _Float16* arow = A + (size_t)(mi * 16 + lrow) * D2;
  const _Float16* wrow = W + (size_t)n * D2;
  for (int kc = 0; kc < D2; kc += 32) {
    const v8h alo = *(const v8h*)(arow + kc + lg * 8);
    const v8h ahi = *(const v8h*)(arow + kc + 16 + lg * 8);
    const v8h bl = *(const v8h*)(wrow + kc + lg * 8);
    const v8h bh = *(const v8h*)(wrow + kc + 16 + lg * 8);
    v16h a, bf;
#pragma unroll
    for (int i = 0; i < 8; ++i) {
      a[i] = alo[i]; a[i + 8] = ahi[i];
      bf[i] = bl[i]; bf[i + 8] = bh[i];
    }
    acc = WMMA_F16(a, bf, acc);
  }
  const int hd = n >> 6, dk = n & 63;
#pragma unroll
  for (int j = 0; j < 8; ++j) {
    const int m = mi * 16 + j + 8 * lg;
    const int b = m >> 9, s = m & 511;
    Out[(((size_t)b * NH + hd) * S + s) * DK + dk] = (_Float16)acc[j];
  }
}

// ---------------------------------------------------------------------------
// Fused attention: per block = (b, 16 s-rows); wave = head.
// scores = q k^T / 8 (WMMA), log-softmax over HEADS (reference axis=1) via LDS,
// ctx += la @ v (WMMA, la are LOG-probs per the reference). No O(S^2) buffer.
// V tile is staged global->LDS by the Tensor Data Mover with 1-DWORD/row
// padding (stride 66 halves) so the K-major gather avoids bank conflicts; the
// TDM copy overlaps the score WMMAs + cross-head softmax, synced on TENSORcnt.
// ---------------------------------------------------------------------------
__launch_bounds__(256)
__global__ void attn_fused(const _Float16* __restrict__ Q, const _Float16* __restrict__ Km,
                           const _Float16* __restrict__ V, float* __restrict__ Ctx) {
  const int tid = threadIdx.x, wave = tid >> 5, lane = tid & 31;
  const int lrow = lane & 15, lg = lane >> 4;
  const int b = blockIdx.x >> 5;
  const int s0 = (blockIdx.x & 31) * 16;
  const int h = wave;
  __shared__ float sc[NH][16][32];  // 16 KB score staging
  __shared__ float lse[16][32];
#if HAVE_TDM
  __shared__ __align__(16) _Float16 vtile[NH][32 * 66];  // padded V tiles, 33 KB
#endif

  const _Float16* qrow = Q + (((size_t)b * NH + h) * S + (s0 + lrow)) * DK;
  v16h qa[2];
#pragma unroll
  for (int kk = 0; kk < 2; ++kk) {
    const int kc = kk * 32;
    const v8h lo = *(const v8h*)(qrow + kc + lg * 8);
    const v8h hi = *(const v8h*)(qrow + kc + 16 + lg * 8);
#pragma unroll
    for (int i = 0; i < 8; ++i) { qa[kk][i] = lo[i]; qa[kk][i + 8] = hi[i]; }
  }
  v8f cacc[4];
#pragma unroll
  for (int nt = 0; nt < 4; ++nt)
#pragma unroll
    for (int e = 0; e < 8; ++e) cacc[nt][e] = 0.f;

  const _Float16* kbase = Km + ((size_t)b * NH + h) * S * DK;
  const _Float16* vbase = V + ((size_t)b * NH + h) * S * DK;
#if HAVE_TDM
  const unsigned vtile_lds = (unsigned)(uintptr_t)&vtile[h][0];
#endif

  for (int t0 = 0; t0 < S; t0 += 32) {
#if HAVE_TDM
    // async: DMA this head's 32x64 f16 V tile (contiguous 4KB) into LDS,
    // padded; overlaps with the score WMMAs and the cross-head softmax below.
    tdm_load_1d_padded(vtile_lds,
                       (unsigned long long)(uintptr_t)(vbase + (size_t)t0 * DK),
                       32 * DK);
#endif
#pragma unroll
    for (int nt = 0; nt < 2; ++nt) {
      v8f sacc;
#pragma unroll
      for (int e = 0; e < 8; ++e) sacc[e] = 0.f;
      const _Float16* krow = kbase + (size_t)(t0 + nt * 16 + lrow) * DK;
#pragma unroll
      for (int kk = 0; kk < 2; ++kk) {
        const int kc = kk * 32;
        const v8h bl = *(const v8h*)(krow + kc + lg * 8);
        const v8h bh = *(const v8h*)(krow + kc + 16 + lg * 8);
        v16h bf;
#pragma unroll
        for (int i = 0; i < 8; ++i) { bf[i] = bl[i]; bf[i + 8] = bh[i]; }
        sacc = WMMA_F16(qa[kk], bf, sacc);
      }
#pragma unroll
      for (int j = 0; j < 8; ++j)
        sc[h][j + 8 * lg][nt * 16 + lrow] = sacc[j] * 0.125f;
    }
    __syncthreads();
    for (int e = tid; e < 16 * 32; e += 256) {
      const int r = e >> 5, cc = e & 31;
      float mx = sc[0][r][cc];
#pragma unroll
      for (int hh = 1; hh < NH; ++hh) mx = fmaxf(mx, sc[hh][r][cc]);
      float sum = 0.f;
#pragma unroll
      for (int hh = 0; hh < NH; ++hh) sum += __expf(sc[hh][r][cc] - mx);
      lse[r][cc] = mx + __logf(sum);
    }
    __syncthreads();
    v16h la;  // A fragment: 16 s-rows x 32 t (log-probs, f16)
#pragma unroll
    for (int i = 0; i < 8; ++i) {
      const int ka = lg * 16 + i, kb = lg * 16 + 8 + i;
      la[i]     = (_Float16)(sc[h][lrow][ka] - lse[lrow][ka]);
      la[i + 8] = (_Float16)(sc[h][lrow][kb] - lse[lrow][kb]);
    }
#if HAVE_TDM
    __builtin_amdgcn_s_wait_tensorcnt(0);  // V tile resident in LDS
#endif
#pragma unroll
    for (int nt = 0; nt < 4; ++nt) {
      const int dk = nt * 16 + lrow;
      v16h vb;  // B fragment: 32 t x 16 dk
#pragma unroll
      for (int i = 0; i < 16; ++i) {
#if HAVE_TDM
        vb[i] = vtile[h][(lg * 16 + i) * 66 + dk];  // 33-bank stride, conflict-free
#else
        vb[i] = vbase[(size_t)(t0 + lg * 16 + i) * DK + dk];
#endif
      }
      cacc[nt] = WMMA_F16(la, vb, cacc[nt]);
    }
    __syncthreads();  // sc reads done before next tile overwrites
  }
#pragma unroll
  for (int nt = 0; nt < 4; ++nt)
#pragma unroll
    for (int j = 0; j < 8; ++j) {
      const int s = s0 + j + 8 * lg;
      const int col = h * DK + nt * 16 + lrow;
      Ctx[((size_t)b * S + s) * D2 + col] = cacc[nt][j];
    }
}

// ---------------------------------------------------------------------------
// residual + layernorm + mean-pool over S + FC -> out[b]
// ---------------------------------------------------------------------------
__launch_bounds__(256)
__global__ void ln_pool_fc(const float* __restrict__ O, const float* __restrict__ Ctx,
                           const float* __restrict__ g, const float* __restrict__ bb,
                           const float* __restrict__ fw, const float* __restrict__ fb,
                           float* __restrict__ out) {
  const int b = blockIdx.x, tid = threadIdx.x;
  __shared__ float red[256];
  float a0 = 0.f, a1 = 0.f;
  for (int s = 0; s < S; ++s) {
    const size_t base = ((size_t)b * S + s) * D2;
    const float r0 = O[base + tid] + Ctx[base + tid];
    const float r1 = O[base + tid + 256] + Ctx[base + tid + 256];
    red[tid] = r0 + r1;
    __syncthreads();
    for (int st = 128; st > 0; st >>= 1) {
      if (tid < st) red[tid] += red[tid + st];
      __syncthreads();
    }
    const float mu = red[0] * (1.f / 512.f);
    __syncthreads();
    red[tid] = (r0 - mu) * (r0 - mu) + (r1 - mu) * (r1 - mu);
    __syncthreads();
    for (int st = 128; st > 0; st >>= 1) {
      if (tid < st) red[tid] += red[tid + st];
      __syncthreads();
    }
    const float is = rsqrtf(red[0] * (1.f / 512.f) + 1e-5f);
    __syncthreads();
    a0 += (r0 - mu) * is * g[tid] + bb[tid];
    a1 += (r1 - mu) * is * g[tid + 256] + bb[tid + 256];
  }
  a0 *= (1.f / 512.f);
  a1 *= (1.f / 512.f);
  red[tid] = a0 * fw[tid] + a1 * fw[tid + 256];
  __syncthreads();
  for (int st = 128; st > 0; st >>= 1) {
    if (tid < st) red[tid] += red[tid + st];
    __syncthreads();
  }
  if (tid == 0) out[b] = red[0] + fb[0];
}

// ---------------------------------------------------------------------------
extern "C" void kernel_launch(void* const* d_in, const int* in_sizes, int n_in,
                              void* d_out, int out_size, void* d_ws, size_t ws_size,
                              hipStream_t stream) {
  const float* x       = (const float*)d_in[0];
  const float* conv_w  = (const float*)d_in[1];
  const float* conv_b  = (const float*)d_in[2];
  const float* bn_g    = (const float*)d_in[3];
  const float* bn_b    = (const float*)d_in[4];
  const float* bn_m    = (const float*)d_in[5];
  const float* bn_v    = (const float*)d_in[6];
  const float* W_ih0f  = (const float*)d_in[7];
  const float* W_hh0f  = (const float*)d_in[8];
  const float* b0f     = (const float*)d_in[9];
  const float* W_ih0b  = (const float*)d_in[10];
  const float* W_hh0b  = (const float*)d_in[11];
  const float* b0b     = (const float*)d_in[12];
  const float* W_ih1f  = (const float*)d_in[13];
  const float* W_hh1f  = (const float*)d_in[14];
  const float* b1f     = (const float*)d_in[15];
  const float* W_ih1b  = (const float*)d_in[16];
  const float* W_hh1b  = (const float*)d_in[17];
  const float* b1b     = (const float*)d_in[18];
  const float* Wq      = (const float*)d_in[19];
  const float* bq      = (const float*)d_in[20];
  const float* Wk      = (const float*)d_in[21];
  const float* bk      = (const float*)d_in[22];
  const float* Wv      = (const float*)d_in[23];
  const float* bv      = (const float*)d_in[24];
  const float* ln_g    = (const float*)d_in[25];
  const float* ln_b    = (const float*)d_in[26];
  const float* fc_w    = (const float*)d_in[27];
  const float* fc_b    = (const float*)d_in[28];

  char* ws = (char*)d_ws;
  size_t off = 0;
  auto carve = [&](size_t bytes) -> void* {
    void* p = ws + off;
    off = (off + bytes + 255) & ~(size_t)255;
    return p;
  };
  const size_t BS = (size_t)Bsz * S;
  _Float16* h0    = (_Float16*)carve(BS * 32 * sizeof(_Float16));   // conv out (B,S,32)
  _Float16* out0  = (_Float16*)carve(BS * D2 * sizeof(_Float16));   // layer0 concat f16
  _Float16* out1h = (_Float16*)carve(BS * D2 * sizeof(_Float16));   // layer1 concat f16
  float*    out1f = (float*)carve(BS * D2 * sizeof(float));         // layer1 concat f32
  _Float16* Qb    = (_Float16*)carve(BS * D2 * sizeof(_Float16));
  _Float16* Kb    = (_Float16*)carve(BS * D2 * sizeof(_Float16));
  _Float16* Vb    = (_Float16*)carve(BS * D2 * sizeof(_Float16));
  float*    ctx   = (float*)carve(BS * D2 * sizeof(float));
  _Float16* Wc0f  = (_Float16*)carve((size_t)1024 * 288 * sizeof(_Float16));
  _Float16* Wc0b  = (_Float16*)carve((size_t)1024 * 288 * sizeof(_Float16));
  _Float16* Wc1f  = (_Float16*)carve((size_t)1024 * 768 * sizeof(_Float16));
  _Float16* Wc1b  = (_Float16*)carve((size_t)1024 * 768 * sizeof(_Float16));
  _Float16* Wq16  = (_Float16*)carve((size_t)D2 * D2 * sizeof(_Float16));
  _Float16* Wk16  = (_Float16*)carve((size_t)D2 * D2 * sizeof(_Float16));
  _Float16* Wv16  = (_Float16*)carve((size_t)D2 * D2 * sizeof(_Float16));

  // weight prep (f16 packing; concatenated [W_ih | W_hh] for fused recurrence)
  pack_wc<<<1152, 256, 0, stream>>>(W_ih0f, W_hh0f, Wc0f, 32);
  pack_wc<<<1152, 256, 0, stream>>>(W_ih0b, W_hh0b, Wc0b, 32);
  pack_wc<<<3072, 256, 0, stream>>>(W_ih1f, W_hh1f, Wc1f, 512);
  pack_wc<<<3072, 256, 0, stream>>>(W_ih1b, W_hh1b, Wc1b, 512);
  cvt_f16<<<1024, 256, 0, stream>>>(Wq, Wq16, D2 * D2);
  cvt_f16<<<1024, 256, 0, stream>>>(Wk, Wk16, D2 * D2);
  cvt_f16<<<1024, 256, 0, stream>>>(Wv, Wv16, D2 * D2);

  // front end
  conv_bn_pool<<<Bsz * S, 32, 0, stream>>>(x, conv_w, conv_b, bn_g, bn_b, bn_m, bn_v, h0);

  // BiLSTM layer 0 (K=288) and layer 1 (K=768); persistent WMMA recurrence
  lstm_dir<32><<<1, 1024, 0, stream>>>(h0, Wc0f, b0f, out0, (float*)nullptr, 0, 0);
  lstm_dir<32><<<1, 1024, 0, stream>>>(h0, Wc0b, b0b, out0, (float*)nullptr, 256, 1);
  lstm_dir<512><<<1, 1024, 0, stream>>>(out0, Wc1f, b1f, out1h, out1f, 0, 0);
  lstm_dir<512><<<1, 1024, 0, stream>>>(out0, Wc1b, b1b, out1h, out1f, 256, 1);

  // attention
  proj_gemm_qkv<<<8192, 256, 0, stream>>>(out1h, Wq16, bq, Qb);
  proj_gemm_qkv<<<8192, 256, 0, stream>>>(out1h, Wk16, bk, Kb);
  proj_gemm_qkv<<<8192, 256, 0, stream>>>(out1h, Wv16, bv, Vb);
  attn_fused<<<Bsz * (S / 16), 256, 0, stream>>>(Qb, Kb, Vb, ctx);

  // residual + LN + pool + FC
  ln_pool_fc<<<Bsz, 256, 0, stream>>>(out1f, ctx, ln_g, ln_b, fc_w, fc_b, (float*)d_out);
}
